// RETAIN_824633721475
// MI455X (gfx1250) — matrix operand order repaired and checked by reading, not compiled, not run on hardware
//
#include <hip/hip_runtime.h>
#include <hip/hip_bf16.h>
#include <math.h>

typedef __attribute__((ext_vector_type(16))) _Float16 v16h;
typedef __attribute__((ext_vector_type(8)))  _Float16 v8h;
typedef __attribute__((ext_vector_type(8)))  float    v8f;

#define B_     32
#define T_     64
#define IN_DIM 4894
#define ROWS   2048   // B*T
#define G3     384    // 3*H

// ---------------------------------------------------------------------------
// WMMA helpers (gfx1250 wave32, 16x16x32 f16 -> f32)
// ---------------------------------------------------------------------------
__device__ __forceinline__ v8f wmma16(v16h a, v16h b, v8f c) {
  return __builtin_amdgcn_wmma_f32_16x16x32_f16(false, a, false, b, (short)0, c,
                                                false, false);
}

// A-tile (16x32) from f32 row-major source, K-guarded (tail chunks only).
// Lane layout (ISA 7.12.2): m = lane&15, khalf = lane>>4;
//   elems 0..7  -> K = khalf*8 + e, elems 8..15 -> K = 16 + khalf*8 + e
__device__ __forceinline__ v16h a_tile_f32(const float* __restrict src, int stride,
                                           int kbase, int kreal, int lane) {
  int m = lane & 15, kh = lane >> 4;
  const float* p = src + (size_t)m * stride + kbase + kh * 8;
  int k0 = kbase + kh * 8;
  v16h a;
#pragma unroll
  for (int e = 0; e < 8; ++e)
    a[e] = (k0 + e < kreal) ? (_Float16)p[e] : (_Float16)0.0f;
#pragma unroll
  for (int e = 0; e < 8; ++e)
    a[8 + e] = (k0 + 16 + e < kreal) ? (_Float16)p[16 + e] : (_Float16)0.0f;
  return a;
}

// Unguarded A-tile: two contiguous 32B float segments per lane -> 4x b128.
__device__ __forceinline__ v16h a_tile_f32_nog(const float* __restrict src,
                                               int stride, int kbase, int lane) {
  int m = lane & 15, kh = lane >> 4;
  const float* p = src + (size_t)m * stride + kbase + kh * 8;
  v16h a;
#pragma unroll
  for (int e = 0; e < 8; ++e) {
    a[e]     = (_Float16)p[e];
    a[8 + e] = (_Float16)p[16 + e];
  }
  return a;
}

// A-tile from f16 LDS (K multiple of 32, no guard): two 16B ds loads per lane.
__device__ __forceinline__ v16h a_tile_h(const _Float16* src, int stride,
                                         int kbase, int lane) {
  int m = lane & 15, kh = lane >> 4;
  const _Float16* p = src + m * stride + kbase + kh * 8;
  v8h lo = *(const v8h*)p;
  v8h hi = *(const v8h*)(p + 16);
  v16h a;
#pragma unroll
  for (int e = 0; e < 8; ++e) { a[e] = lo[e]; a[8 + e] = hi[e]; }
  return a;
}

// Packed B-tile: contiguous 32 bytes per lane (pre-swizzled by k_pack).
__device__ __forceinline__ v16h b_tile(const _Float16* __restrict packed, int kc,
                                       int NT, int nt, int lane) {
  return *(const v16h*)(packed + ((size_t)((kc * NT + nt) * 32 + lane)) * 16);
}

// ---------------------------------------------------------------------------
// Pack W[N x Kstride] (row-major, we compute X @ W.T) into f16 WMMA B-tile
// layout: dst[((kc*NT+nt)*32+lane)*16+e] = W[nt*16+lane%16][kc*32+(lane/16)*8+(e<8?e:8+e)]
// ---------------------------------------------------------------------------
__global__ void k_pack(const float* __restrict W, _Float16* __restrict dst,
                       int Kstride, int Kreal, int NT) {
  int tile = blockIdx.x;
  int nt = tile % NT;
  int lane = threadIdx.x;
  int n = nt * 16 + (lane & 15);
  int kb = (tile / NT) * 32 + (lane >> 4) * 8;
  _Float16* o = dst + (size_t)tile * 512 + lane * 16;
#pragma unroll
  for (int e = 0; e < 16; ++e) {
    int k2 = kb + (e < 8 ? e : 8 + e);
    o[e] = (k2 < Kreal) ? (_Float16)W[(size_t)n * Kstride + k2] : (_Float16)0.0f;
  }
}

// ---------------------------------------------------------------------------
// emb = x @ W_emb.T   [2048 x 4894] @ [4894 x 128] -> f32 [2048 x 128]
// HBM-bound stage (40 MB of x): split-K across the 8 waves so each K-chunk of
// x is fetched exactly once per block; software prefetch one wave-iteration
// ahead; cross-wave partial reduction via LDS float atomics (ds_add_f32).
// Only chunk 152 (cols 4864..4893) needs the K guard.
// ---------------------------------------------------------------------------
__global__ __launch_bounds__(256) void k_emb(const float* __restrict x,
                                             const _Float16* __restrict pW,
                                             float* __restrict emb) {
  __shared__ float red[16][128];
  const int tid = threadIdx.x, lane = tid & 31, wid = tid >> 5;
  const int rbase = blockIdx.x * 16;
  const int hl = lane >> 4, ln = lane & 15;
  const int m = lane & 15;
  const float* xa = x + (size_t)rbase * IN_DIM;

#pragma unroll
  for (int e = 0; e < 8; ++e) {
    int idx = e * 256 + tid;
    red[idx >> 7][idx & 127] = 0.f;
  }
  __syncthreads();

  v8f acc[8] = {{}, {}, {}, {}, {}, {}, {}, {}};
  for (int kc = wid; kc < 152; kc += 8) {
    int pf = kc + 8; if (pf > 152) pf = 152;   // prefetch next owned chunk
    __builtin_prefetch(xa + (size_t)m * IN_DIM + pf * 32 + hl * 8, 0, 0);
    v16h a = a_tile_f32_nog(xa, IN_DIM, kc * 32, lane);
#pragma unroll
    for (int nt = 0; nt < 8; ++nt)
      acc[nt] = wmma16(a, b_tile(pW, kc, 8, nt, lane), acc[nt]);
  }
  if (wid == 0) {  // guarded tail chunk (uniform per wave -> EXEC stays all-1)
    v16h a = a_tile_f32(xa, IN_DIM, 152 * 32, IN_DIM, lane);
#pragma unroll
    for (int nt = 0; nt < 8; ++nt)
      acc[nt] = wmma16(a, b_tile(pW, 152, 8, nt, lane), acc[nt]);
  }
  // cross-wave reduction of split-K partials
#pragma unroll
  for (int nt = 0; nt < 8; ++nt)
#pragma unroll
    for (int v = 0; v < 8; ++v)
      atomicAdd(&red[v + 8 * hl][nt * 16 + ln], acc[nt][v]);
  __syncthreads();
#pragma unroll
  for (int e = 0; e < 8; ++e) {
    int idx = e * 256 + tid; int r = idx >> 7, c = idx & 127;
    emb[(size_t)(rbase + r) * 128 + c] = red[r][c];
  }
}

// ---------------------------------------------------------------------------
// Hoisted input gates for both GRUs:
//   G{a,b}[row, 0:384] = temb[row] @ Wih.T + bih,  temb = [emb(row), t(row)]
// The 129th feature (time) is a rank-1 epilogue on the WMMA accumulators.
// ---------------------------------------------------------------------------
__global__ __launch_bounds__(256) void k_gates(
    const float* __restrict emb, const float* __restrict t,
    const float* __restrict WihA, const float* __restrict WihB,
    const float* __restrict bihA, const float* __restrict bihB,
    const _Float16* __restrict pWihA, const _Float16* __restrict pWihB,
    float* __restrict Ga, float* __restrict Gb) {
  const int tid = threadIdx.x, lane = tid & 31, wid = tid >> 5;
  const int rbase = blockIdx.x * 16;
  const int hl = lane >> 4, ln = lane & 15;
  v8f aA[3] = {{}, {}, {}}, aB[3] = {{}, {}, {}};
  const float* ea = emb + (size_t)rbase * 128;
#pragma unroll
  for (int kc = 0; kc < 4; ++kc) {
    v16h a = a_tile_f32_nog(ea, 128, kc * 32, lane);
#pragma unroll
    for (int j = 0; j < 3; ++j) {
      aA[j] = wmma16(a, b_tile(pWihA, kc, 24, wid * 3 + j, lane), aA[j]);
      aB[j] = wmma16(a, b_tile(pWihB, kc, 24, wid * 3 + j, lane), aB[j]);
    }
  }
#pragma unroll
  for (int j = 0; j < 3; ++j) {
    int n = (wid * 3 + j) * 16 + ln;
    float wA = WihA[(size_t)n * 129 + 128], wB = WihB[(size_t)n * 129 + 128];
    float bA_ = bihA[n], bB_ = bihB[n];
#pragma unroll
    for (int v = 0; v < 8; ++v) {
      int row = rbase + v + 8 * hl;
      float tv = t[row];
      Ga[(size_t)row * G3 + n] = aA[j][v] + tv * wA + bA_;
      Gb[(size_t)row * G3 + n] = aB[j][v] + tv * wB + bB_;
    }
  }
}

// ---------------------------------------------------------------------------
// Fused scan: per block, 16 (b,i) rows; loop k = 0..63:
//   gh = h @ Whh.T + bhh  (WMMA, weight-stationary B-tiles in VGPRs)
//   GRU update (r,z,n), attention score, beta = hb @ W_beta.T,
//   online-softmax accumulation of c = sum_k alpha_k * tanh(beta_k) * emb[i-k].
// Finalize: logits = (mask ? c . W_out : 0) + b_out.
// ---------------------------------------------------------------------------
__global__ __launch_bounds__(256) void k_scan(
    const float* __restrict emb, const float* __restrict Ga,
    const float* __restrict Gb, const _Float16* __restrict pWhhA,
    const _Float16* __restrict pWhhB, const _Float16* __restrict pWbeta,
    const float* __restrict bihA, const float* __restrict bihB,
    const float* __restrict bhhA, const float* __restrict bhhB,
    const float* __restrict w_alpha, const float* __restrict b_alpha,
    const float* __restrict b_beta, const float* __restrict W_out,
    const float* __restrict b_out, const int* __restrict lengths,
    float* __restrict out) {
  __shared__ float    hA[16][128];
  __shared__ float    hB[16][128];
  __shared__ _Float16 hf16[16][128];
  __shared__ float    gh[16][G3];
  __shared__ float    betaS[16][128];
  __shared__ float    accS[16][128];
  __shared__ float    mS[16], dS[16], sS[16], coefS[16], wS[16];

  const int tid = threadIdx.x;
  const int lane = tid & 31, wid = tid >> 5;
  const int rbase = blockIdx.x * 16;
  const int bidx = rbase / T_;
  const int ibase = rbase % T_;
  const int hl = lane >> 4, ln = lane & 15;

#pragma unroll
  for (int e = 0; e < 8; ++e) {
    int idx = e * 256 + tid; int r = idx >> 7, c = idx & 127;
    hA[r][c] = 0.f; hB[r][c] = 0.f; accS[r][c] = 0.f;
  }
  if (tid < 16) { mS[tid] = -1e30f; dS[tid] = 0.f; }
  __syncthreads();

  // Weight-stationary: hold all recurrent + beta B-tiles in VGPRs for 64 steps.
  v16h bA[3][4], bBt[3][4], bBe[4];
#pragma unroll
  for (int j = 0; j < 3; ++j)
#pragma unroll
    for (int kc = 0; kc < 4; ++kc) {
      bA[j][kc]  = b_tile(pWhhA, kc, 24, wid * 3 + j, lane);
      bBt[j][kc] = b_tile(pWhhB, kc, 24, wid * 3 + j, lane);
    }
#pragma unroll
  for (int kc = 0; kc < 4; ++kc) bBe[kc] = b_tile(pWbeta, kc, 8, wid, lane);

  for (int k = 0; k < T_; ++k) {
    // ---- GRU A: stage h -> f16, gh = h @ WhhA.T + bhhA ----
#pragma unroll
    for (int e = 0; e < 8; ++e) { int idx = e * 256 + tid;
      hf16[idx >> 7][idx & 127] = (_Float16)hA[idx >> 7][idx & 127]; }
    __syncthreads();
    {
      v8f acc[3] = {{}, {}, {}};
#pragma unroll
      for (int kc = 0; kc < 4; ++kc) {
        v16h a = a_tile_h(&hf16[0][0], 128, kc * 32, lane);
#pragma unroll
        for (int j = 0; j < 3; ++j) acc[j] = wmma16(a, bA[j][kc], acc[j]);
      }
      __syncthreads();
#pragma unroll
      for (int j = 0; j < 3; ++j) {
        int n = (wid * 3 + j) * 16 + ln;
#pragma unroll
        for (int v = 0; v < 8; ++v) gh[v + 8 * hl][n] = acc[j][v] + bhhA[n];
      }
    }
    __syncthreads();
    // GRU A elementwise update
#pragma unroll
    for (int e = 0; e < 8; ++e) {
      int idx = e * 256 + tid; int r = idx >> 7, c = idx & 127;
      bool ug = (k <= ibase + r);
      int grow = rbase + r - k;
      float gir = ug ? Ga[(size_t)grow * G3 + c]       : bihA[c];
      float giz = ug ? Ga[(size_t)grow * G3 + 128 + c] : bihA[128 + c];
      float gin = ug ? Ga[(size_t)grow * G3 + 256 + c] : bihA[256 + c];
      float rg = 1.f / (1.f + __expf(-(gir + gh[r][c])));
      float zg = 1.f / (1.f + __expf(-(giz + gh[r][128 + c])));
      float ng = tanhf(gin + rg * gh[r][256 + c]);
      hA[r][c] = (1.f - zg) * ng + zg * hA[r][c];
    }
    __syncthreads();
    // attention score on new hA; concurrently stage hB(old) for its GEMM
    if (tid < 16) {
      float s = b_alpha[0];
      for (int c = 0; c < 128; ++c) s += hA[tid][c] * w_alpha[c];
      sS[tid] = s;
    }
#pragma unroll
    for (int e = 0; e < 8; ++e) { int idx = e * 256 + tid;
      hf16[idx >> 7][idx & 127] = (_Float16)hB[idx >> 7][idx & 127]; }
    __syncthreads();
    // ---- GRU B ----
    {
      v8f acc[3] = {{}, {}, {}};
#pragma unroll
      for (int kc = 0; kc < 4; ++kc) {
        v16h a = a_tile_h(&hf16[0][0], 128, kc * 32, lane);
#pragma unroll
        for (int j = 0; j < 3; ++j) acc[j] = wmma16(a, bBt[j][kc], acc[j]);
      }
      __syncthreads();
#pragma unroll
      for (int j = 0; j < 3; ++j) {
        int n = (wid * 3 + j) * 16 + ln;
#pragma unroll
        for (int v = 0; v < 8; ++v) gh[v + 8 * hl][n] = acc[j][v] + bhhB[n];
      }
    }
    __syncthreads();
#pragma unroll
    for (int e = 0; e < 8; ++e) {
      int idx = e * 256 + tid; int r = idx >> 7, c = idx & 127;
      bool ug = (k <= ibase + r);
      int grow = rbase + r - k;
      float gir = ug ? Gb[(size_t)grow * G3 + c]       : bihB[c];
      float giz = ug ? Gb[(size_t)grow * G3 + 128 + c] : bihB[128 + c];
      float gin = ug ? Gb[(size_t)grow * G3 + 256 + c] : bihB[256 + c];
      float rg = 1.f / (1.f + __expf(-(gir + gh[r][c])));
      float zg = 1.f / (1.f + __expf(-(giz + gh[r][128 + c])));
      float ng = tanhf(gin + rg * gh[r][256 + c]);
      hB[r][c] = (1.f - zg) * ng + zg * hB[r][c];
    }
    __syncthreads();
    // beta = hB(new) @ W_beta.T + b_beta
#pragma unroll
    for (int e = 0; e < 8; ++e) { int idx = e * 256 + tid;
      hf16[idx >> 7][idx & 127] = (_Float16)hB[idx >> 7][idx & 127]; }
    __syncthreads();
    {
      v8f acc = {};
#pragma unroll
      for (int kc = 0; kc < 4; ++kc) {
        v16h a = a_tile_h(&hf16[0][0], 128, kc * 32, lane);
        acc = wmma16(a, bBe[kc], acc);
      }
      int n = wid * 16 + ln;
#pragma unroll
      for (int v = 0; v < 8; ++v) betaS[v + 8 * hl][n] = acc[v] + b_beta[n];
    }
    __syncthreads();
    // online-softmax coefficients (per row)
    if (tid < 16) {
      if (k <= ibase + tid) {
        float s = sS[tid], mo = mS[tid];
        float mn = fmaxf(mo, s);
        float cf = __expf(mo - mn);
        float w  = __expf(s - mn);
        dS[tid] = dS[tid] * cf + w;
        mS[tid] = mn; coefS[tid] = cf; wS[tid] = w;
      } else { coefS[tid] = 1.f; wS[tid] = 0.f; }
    }
    __syncthreads();
    // streaming context accumulation: acc = acc*cf + w * tanh(beta) * emb[i-k]
#pragma unroll
    for (int e = 0; e < 8; ++e) {
      int idx = e * 256 + tid; int r = idx >> 7, c = idx & 127;
      float ev = (k <= ibase + r) ? emb[(size_t)(rbase + r - k) * 128 + c] : 0.f;
      accS[r][c] = accS[r][c] * coefS[r] + wS[r] * tanhf(betaS[r][c]) * ev;
    }
    __syncthreads();
  }

  // finalize: c = acc/d, row-length mask, logits = c . W_out + b_out
  if (tid < 16) {
    int r = tid;
    float inv = 1.f / dS[r];
    float lg = 0.f;
    for (int c = 0; c < 128; ++c) lg += accS[r][c] * inv * W_out[c];
    bool rv = (ibase + r < lengths[bidx]);
    out[rbase + r] = (rv ? lg : 0.f) + b_out[0];
  }
}

// ---------------------------------------------------------------------------
extern "C" void kernel_launch(void* const* d_in, const int* in_sizes, int n_in,
                              void* d_out, int out_size, void* d_ws, size_t ws_size,
                              hipStream_t stream) {
  const float* x       = (const float*)d_in[0];
  const float* t       = (const float*)d_in[1];
  const int*   lens    = (const int*)d_in[2];
  const float* W_emb   = (const float*)d_in[3];
  const float* Wih_a   = (const float*)d_in[4];
  const float* Whh_a   = (const float*)d_in[5];
  const float* bih_a   = (const float*)d_in[6];
  const float* bhh_a   = (const float*)d_in[7];
  const float* Wih_b   = (const float*)d_in[8];
  const float* Whh_b   = (const float*)d_in[9];
  const float* bih_b   = (const float*)d_in[10];
  const float* bhh_b   = (const float*)d_in[11];
  const float* w_alpha = (const float*)d_in[12];
  const float* b_alpha = (const float*)d_in[13];
  const float* W_beta  = (const float*)d_in[14];
  const float* b_beta  = (const float*)d_in[15];
  const float* W_out   = (const float*)d_in[16];
  const float* b_out   = (const float*)d_in[17];
  float* out = (float*)d_out;

  // workspace layout (~9 MB, all 32B aligned)
  float* emb = (float*)d_ws;                               // 2048*128 f32
  float* Ga  = emb + (size_t)ROWS * 128;                   // 2048*384 f32
  float* Gb  = Ga + (size_t)ROWS * G3;                     // 2048*384 f32
  _Float16* pWemb = (_Float16*)(Gb + (size_t)ROWS * G3);   // 153*8*512 f16
  _Float16* pWihA = pWemb + (size_t)153 * 8 * 512;         // 4*24*512 f16
  _Float16* pWihB = pWihA + (size_t)4 * 24 * 512;
  _Float16* pWhhA = pWihB + (size_t)4 * 24 * 512;
  _Float16* pWhhB = pWhhA + (size_t)4 * 24 * 512;
  _Float16* pWbeta = pWhhB + (size_t)4 * 24 * 512;         // 4*8*512 f16

  k_pack<<<153 * 8, 32, 0, stream>>>(W_emb, pWemb, IN_DIM, IN_DIM, 8);
  k_pack<<<4 * 24, 32, 0, stream>>>(Wih_a, pWihA, 129, 128, 24);
  k_pack<<<4 * 24, 32, 0, stream>>>(Wih_b, pWihB, 129, 128, 24);
  k_pack<<<4 * 24, 32, 0, stream>>>(Whh_a, pWhhA, 128, 128, 24);
  k_pack<<<4 * 24, 32, 0, stream>>>(Whh_b, pWhhB, 128, 128, 24);
  k_pack<<<4 * 8, 32, 0, stream>>>(W_beta, pWbeta, 128, 128, 8);

  k_emb<<<ROWS / 16, 256, 0, stream>>>(x, pWemb, emb);
  k_gates<<<ROWS / 16, 256, 0, stream>>>(emb, t, Wih_a, Wih_b, bih_a, bih_b,
                                         pWihA, pWihB, Ga, Gb);
  k_scan<<<ROWS / 16, 256, 0, stream>>>(emb, Ga, Gb, pWhhA, pWhhB, pWbeta,
                                        bih_a, bih_b, bhh_a, bhh_b, w_alpha,
                                        b_alpha, b_beta, W_out, b_out, lens, out);
}
